// StyleLayer_7945689498060
// MI455X (gfx1250) — matrix operand order repaired and checked by reading, not compiled, not run on hardware
//
#include <hip/hip_runtime.h>
#include <hip/hip_bf16.h>
#include <math.h>
#include <stdint.h>

// ---------------------------------------------------------------------------
// Problem constants (from reference)
// ---------------------------------------------------------------------------
#define BATCH 8
#define CIN   512
#define COUT  512
#define SDIM  512
#define HH    64
#define WW    64
#define NTOT  4356          // 66*66 conv output spatial
#define OH    66

// WMMA tiling
#define MTILE 128
#define NTILE 64
#define KTILE 32
#define ASTRIDE 40          // LDS row stride in bf16 elems (64B data + 16B pad)
#define BSTRIDE 40
#define NKC   (CIN / KTILE) // 16 k-chunks per tap

typedef __bf16 v16bf __attribute__((ext_vector_type(16)));
typedef __bf16 v8bf  __attribute__((ext_vector_type(8)));
typedef float  v8f   __attribute__((ext_vector_type(8)));
typedef unsigned int u32x4 __attribute__((ext_vector_type(4)));
typedef unsigned int u32x8 __attribute__((ext_vector_type(8)));

__device__ __forceinline__ unsigned short f2bf(float f) {
    unsigned int u = __float_as_uint(f);
    unsigned int r = u + 0x7fffu + ((u >> 16) & 1u);   // round-to-nearest-even
    return (unsigned short)(r >> 16);
}

// ---------------------------------------------------------------------------
// Workspace layout (bytes)
// ---------------------------------------------------------------------------
constexpr size_t OFF_S     = 0;
constexpr size_t OFF_WSQ   = 16384;
constexpr size_t OFF_DCOEF = OFF_WSQ + 1048576;
constexpr size_t OFF_WBF   = OFF_DCOEF + 16384;
constexpr size_t OFF_XBF   = OFF_WBF + (size_t)BATCH * 9 * COUT * CIN * 2;
constexpr size_t OFF_YCONV = OFF_XBF + (size_t)BATCH * HH * WW * CIN * 2;

// ---------------------------------------------------------------------------
// P1: s[b][ci] = (w[b,:] . affw[ci,:]) / sqrt(SDIM) + affb[ci]
// ---------------------------------------------------------------------------
__global__ __launch_bounds__(256) void style_kernel(
    const float* __restrict__ w, const float* __restrict__ affw,
    const float* __restrict__ affb, float* __restrict__ s)
{
    int idx = blockIdx.x * 256 + threadIdx.x;        // 0..4095
    int b = idx >> 9, ci = idx & 511;
    const float* wr = w + (size_t)b * SDIM;
    const float* ar = affw + (size_t)ci * SDIM;
    float acc = 0.0f;
    for (int k = 0; k < SDIM; ++k) acc = fmaf(wr[k], ar[k], acc);
    s[idx] = acc * 0.04419417382415922f + affb[ci];  // 1/sqrt(512)
}

// ---------------------------------------------------------------------------
// P2: wsq[o][i] = sum over 9 taps of conv_weight^2
// ---------------------------------------------------------------------------
__global__ __launch_bounds__(256) void wsq_kernel(
    const float* __restrict__ cw, float* __restrict__ wsq)
{
    int idx = blockIdx.x * 256 + threadIdx.x;        // 0..262143
    const float* p = cw + (size_t)idx * 9;
    float acc = 0.0f;
    #pragma unroll
    for (int t = 0; t < 9; ++t) acc = fmaf(p[t], p[t], acc);
    wsq[idx] = acc;
}

// ---------------------------------------------------------------------------
// P3: dcoef[b][o] = c2 * input_gain * rsqrt(c2^2 * sum_i s^2*wsq + 1e-8)
// ---------------------------------------------------------------------------
__global__ __launch_bounds__(256) void demod_kernel(
    const float* __restrict__ s, const float* __restrict__ wsq,
    const float* __restrict__ ema, float* __restrict__ dcoef)
{
    int idx = blockIdx.x * 256 + threadIdx.x;        // 0..4095
    int b = idx >> 9, o = idx & 511;
    const float* sr = s + (size_t)b * CIN;
    const float* qr = wsq + (size_t)o * CIN;
    float acc = 0.0f;
    for (int i = 0; i < CIN; ++i) { float sv = sr[i]; acc = fmaf(sv * sv, qr[i], acc); }
    const float c2 = 0.014731391274719738f;          // 1/sqrt(4608)
    float ig = rsqrtf(ema[0]);
    dcoef[idx] = c2 * ig * rsqrtf(c2 * c2 * acc + 1e-8f);
}

// ---------------------------------------------------------------------------
// P4: wbf[b][tap][o][i] = bf16( cw[o][i][tap] * s[b][i] * dcoef[b][o] )
// ---------------------------------------------------------------------------
__global__ __launch_bounds__(256) void modw_kernel(
    const float* __restrict__ cw, const float* __restrict__ s,
    const float* __restrict__ dcoef, unsigned short* __restrict__ wbf)
{
    int idx = blockIdx.x * 256 + threadIdx.x;        // 0..18874367 exactly
    int i  = idx & 511;
    int o  = (idx >> 9) & 511;
    int bt = idx >> 18;                              // 0..71
    int t  = bt % 9;
    int b  = bt / 9;
    float v = cw[((size_t)o * CIN + i) * 9 + t] * s[(size_t)b * CIN + i]
            * dcoef[(size_t)b * COUT + o];
    wbf[idx] = f2bf(v);
}

// ---------------------------------------------------------------------------
// P5: x NCHW f32 -> NHWC bf16 via LDS transpose tile (64ci x 64w per block)
// ---------------------------------------------------------------------------
__global__ __launch_bounds__(256) void xcvt_kernel(
    const float* __restrict__ x, unsigned short* __restrict__ xbf)
{
    __shared__ float tile[64][65];
    int tid = threadIdx.x;
    int bh = blockIdx.x;                 // 0..511
    int b = bh >> 6, h = bh & 63;
    int ci0 = blockIdx.y * 64;
    int sub = tid >> 6;                  // 0..3
    int lnx = tid & 63;
    #pragma unroll
    for (int r = 0; r < 16; ++r) {
        int cc = r * 4 + sub;
        tile[cc][lnx] = x[(((size_t)b * CIN + ci0 + cc) * HH + h) * WW + lnx];
    }
    __syncthreads();
    #pragma unroll
    for (int r = 0; r < 16; ++r) {
        int ww = r * 4 + sub;
        xbf[(((size_t)b * HH + h) * WW + ww) * CIN + ci0 + lnx] = f2bf(tile[lnx][ww]);
    }
}

// ---------------------------------------------------------------------------
// Main conv as implicit GEMM, bf16 WMMA, f32 accumulate.
// A tile (128x32 bf16, regular 2D) staged by the Tensor Data Mover
// (tensor_load_to_lds, TENSORcnt, LDS padding 64B+16B -> ASTRIDE 40).
// B tile (irregular im2col) staged per-lane with GLOBAL_LOAD_ASYNC_TO_LDS_B128
// (ASYNCcnt). Double-buffered: stage k+1 in flight while stage k feeds
// v_wmma_f32_16x16x32_bf16.
// WG: 256 thr (8 waves), tile 128(M) x 64(N); each wave 64M x 16N (4 WMMA/step).
// ---------------------------------------------------------------------------
__global__ __launch_bounds__(256) void conv_wmma_kernel(
    const unsigned short* __restrict__ wbf,
    const unsigned short* __restrict__ xbf,
    const float* __restrict__ bias,
    float* __restrict__ yconv)
{
    __shared__ unsigned short smA[2][MTILE * ASTRIDE];
    __shared__ unsigned short smB[2][NTILE * BSTRIDE];
    __shared__ unsigned short smDump[8];   // sink for invalid-lane async writes

    const int tid   = threadIdx.x;
    const int b     = blockIdx.z;
    const int mBase = blockIdx.y * MTILE;
    const int nBase = blockIdx.x * NTILE;

    const int wave = tid >> 5;
    const int lane = tid & 31;
    const int wm   = wave >> 2;          // 0..1
    const int wn   = wave & 3;           // 0..3
    const int half = lane >> 4;          // 0/1
    const int r16  = lane & 15;

    const v8f zero8 = {0.f,0.f,0.f,0.f,0.f,0.f,0.f,0.f};
    v8f acc[4] = {zero8, zero8, zero8, zero8};

    // cooperative-load roles for B: 4 quads of 16B cover one 32-bf16 row
    const int quad = tid & 3;
    const int row0 = tid >> 2;           // 0..63

    // LDS byte addresses (generic-pointer low 32 bits == LDS offset)
    unsigned ldsAbase[2], ldsB[2];
    #pragma unroll
    for (int bb = 0; bb < 2; ++bb) {
        ldsAbase[bb] = (unsigned)(size_t)&smA[bb][0];
        ldsB[bb]     = (unsigned)(size_t)&smB[bb][row0 * BSTRIDE + quad * 8];
    }
    const unsigned ldsDump = (unsigned)(size_t)&smDump[0];

    // ---- TDM D# group1 (stage-invariant): 2D tile 32(el) x 128(rows),
    // data_size=2B, row stride 512 el, LDS pad: every 16 DWORDs add 4 DWORDs.
    u32x8 g1;
    g1[0] = (1u << 16)        // data_size = 1 -> 2 bytes
          | (1u << 20)        // pad_enable
          | (3u << 22)        // pad_interval = 16 DWORDs (64B)
          | (3u << 25);       // pad_amount   = 4 DWORDs (16B)
    g1[1] = (unsigned)CIN  << 16;   // tensor_dim0 = 512 (bits 79:48 low half)
    g1[2] = (unsigned)COUT << 16;   // tensor_dim1 = 512 (bits 111:80 low half)
    g1[3] = (unsigned)KTILE << 16;  // tile_dim0 = 32 (bits 127:112)
    g1[4] = (unsigned)MTILE;        // tile_dim1 = 128 (bits 143:128)
    g1[5] = (unsigned)CIN;          // tensor_dim0_stride = 512 (bits 207:160)
    g1[6] = 0u;
    g1[7] = 0u;

    // B-tile spatial decode (one n per thread, tap-invariant pieces)
    int nIdx = nBase + row0;
    int oh = nIdx / OH;
    int ow = nIdx - oh * OH;
    bool nValid = (nIdx < NTOT);

    for (int tap = 0; tap < 9; ++tap) {
        int ty = tap / 3, tx = tap - ty * 3;
        int ih = oh + ty - 2;
        int iw = ow + tx - 2;
        bool pixValid = nValid && ((unsigned)ih < (unsigned)HH) && ((unsigned)iw < (unsigned)WW);
        // clamped (always-legal) source address; invalid lanes dump to smDump
        int ihc = min(max(ih, 0), HH - 1);
        int iwc = min(max(iw, 0), WW - 1);
        const unsigned short* wsrcTapBase =
            wbf + (((size_t)(b * 9 + tap) * COUT + mBase) * CIN);
        const unsigned short* xsrcTap =
            xbf + (((size_t)b * HH + ihc) * WW + iwc) * CIN + quad * 8;
        unsigned bSlot0 = pixValid ? ldsB[0] : ldsDump;
        unsigned bSlot1 = pixValid ? ldsB[1] : ldsDump;

        // zero the never-loaded (zero-pad) B slots in both buffers, once per tap
        __syncthreads();                 // previous tap fully consumed
        if (!pixValid) {
            uint4 z = {0u, 0u, 0u, 0u};
            *reinterpret_cast<uint4*>(&smB[0][row0 * BSTRIDE + quad * 8]) = z;
            *reinterpret_cast<uint4*>(&smB[1][row0 * BSTRIDE + quad * 8]) = z;
        }
        __syncthreads();

        // ---- prologue: stage kc=0 into buffer 0 ----
        if (wave == 0) {
            unsigned long long ga = (unsigned long long)(size_t)wsrcTapBase;
            u32x4 g0;
            g0[0] = 1u;                                  // count = 1 valid desc
            g0[1] = ldsAbase[0];                         // lds_addr
            g0[2] = (unsigned)ga;                        // global_addr[31:0]
            g0[3] = (unsigned)(ga >> 32) | 0x80000000u;  // addr[56:32] | type=2
            asm volatile("tensor_load_to_lds %0, %1"
                         :: "s"(g0), "s"(g1) : "memory");
        }
        asm volatile("global_load_async_to_lds_b128 %0, %1, off"
                     :: "v"(bSlot0), "v"(xsrcTap) : "memory");

        for (int kc = 0; kc < NKC; ++kc) {
            const int cur = kc & 1;
            // ---- prefetch stage kc+1 into the other buffer ----
            if (kc < NKC - 1) {
                if (wave == 0) {
                    unsigned long long ga = (unsigned long long)(size_t)
                        (wsrcTapBase + (kc + 1) * KTILE);
                    u32x4 g0;
                    g0[0] = 1u;
                    g0[1] = cur ? ldsAbase[0] : ldsAbase[1];
                    g0[2] = (unsigned)ga;
                    g0[3] = (unsigned)(ga >> 32) | 0x80000000u;
                    asm volatile("tensor_load_to_lds %0, %1"
                                 :: "s"(g0), "s"(g1) : "memory");
                }
                {
                    const unsigned short* bsrc = xsrcTap + (kc + 1) * KTILE;
                    unsigned dB = cur ? bSlot0 : bSlot1;
                    asm volatile("global_load_async_to_lds_b128 %0, %1, off"
                                 :: "v"(dB), "v"(bsrc) : "memory");
                }
                // retire current stage; one newer op of each kind stays in flight
                asm volatile("s_wait_asynccnt 0x1" ::: "memory");
                __builtin_amdgcn_s_wait_tensorcnt(1);
            } else {
                asm volatile("s_wait_asynccnt 0x0" ::: "memory");
                __builtin_amdgcn_s_wait_tensorcnt(0);
            }
            __syncthreads();             // stage visible to all waves

            // ---- WMMA on current buffer ----
            const unsigned short* bptr =
                &smB[cur][(wn * 16 + r16) * BSTRIDE + half * 8];
            v8bf blo = *reinterpret_cast<const v8bf*>(bptr);
            v8bf bhi = *reinterpret_cast<const v8bf*>(bptr + 16);
            v16bf bfrag = __builtin_shufflevector(blo, bhi,
                0,1,2,3,4,5,6,7,8,9,10,11,12,13,14,15);

            #pragma unroll
            for (int s4 = 0; s4 < 4; ++s4) {
                const unsigned short* aptr =
                    &smA[cur][(wm * 64 + s4 * 16 + r16) * ASTRIDE + half * 8];
                v8bf alo = *reinterpret_cast<const v8bf*>(aptr);
                v8bf ahi = *reinterpret_cast<const v8bf*>(aptr + 16);
                v16bf afrag = __builtin_shufflevector(alo, ahi,
                    0,1,2,3,4,5,6,7,8,9,10,11,12,13,14,15);
                acc[s4] = __builtin_amdgcn_wmma_f32_16x16x32_bf16(
                    false, afrag, false, bfrag, (short)0, acc[s4], false, false);
            }
            __syncthreads();             // done reading before buffer reuse
        }
    }

    // store: C/D layout — vgpr j, lanes0-15: (M=j, N=lane); lanes16-31: (M=8+j)
    int nOut = nBase + wn * 16 + r16;
    if (nOut < NTOT) {
        #pragma unroll
        for (int s4 = 0; s4 < 4; ++s4) {
            #pragma unroll
            for (int j = 0; j < 8; ++j) {
                int m = mBase + wm * 64 + s4 * 16 + j + 8 * half;
                yconv[((size_t)(b * COUT + m)) * NTOT + nOut] = acc[s4][j] + bias[m];
            }
        }
    }
}

// ---------------------------------------------------------------------------
// Fused filtered-leaky-relu: up2 + pad(9,8) + 12tap(x) + 12tap(y) [gain 2 each]
// -> lrelu*sqrt2, clamp 256 -> 12tap(x) + 12tap(y) + down2, per 16x16 out tile.
// ---------------------------------------------------------------------------
__global__ __launch_bounds__(256) void filter_fused_kernel(
    const float* __restrict__ yconv, const float* __restrict__ upf,
    const float* __restrict__ dnf, float* __restrict__ out)
{
    __shared__ float fu[12], fd[12];
    __shared__ float sy[27][28];     // y patch
    __shared__ float st1[27][44];    // horizontally up-filtered rows
    __shared__ float sz[42][43];     // activated upsampled tile
    __shared__ float sA[42][17];     // horizontally down-filtered

    const int tid = threadIdx.x;
    const int bc  = blockIdx.z;                 // b*512 + c
    const int j0  = blockIdx.x * 16;
    const int i0  = blockIdx.y * 16;
    const int vmin = 2 * i0, umin = 2 * j0;

    if (tid < 12) {
        fu[tid] = upf[11 - tid] * 2.0f;         // reversed, gain sqrt(4) per pass
        fd[tid] = dnf[11 - tid];                // reversed, gain 1
    }

    int q0 = (vmin <= 9) ? 0 : ((vmin - 8) >> 1);
    int q1 = (vmin + 43) >> 1; if (q1 > 65) q1 = 65;
    int NQ = q1 - q0 + 1;                       // <= 27
    int p0 = (umin <= 9) ? 0 : ((umin - 8) >> 1);
    int p1 = (umin + 43) >> 1; if (p1 > 65) p1 = 65;
    int NP = p1 - p0 + 1;

    const float* img = yconv + (size_t)bc * NTOT;

    for (int idx = tid; idx < 27 * 27; idx += 256) {
        int qi = idx / 27, pi = idx - qi * 27;
        float v = 0.0f;
        if (qi < NQ && pi < NP) v = img[(q0 + qi) * OH + (p0 + pi)];
        sy[qi][pi] = v;
    }
    __syncthreads();

    for (int idx = tid; idx < 27 * 42; idx += 256) {
        int qi = idx / 42, ui = idx - qi * 42;
        int u = umin + ui;
        float a = 0.0f;
        if (qi < NQ) {
            #pragma unroll
            for (int t = 0; t < 12; ++t) {
                int j = u + t - 9;
                if (j >= 0 && j < 132 && ((j & 1) == 0))
                    a = fmaf(fu[t], sy[qi][(j >> 1) - p0], a);
            }
        }
        st1[qi][ui] = a;
    }
    __syncthreads();

    for (int idx = tid; idx < 42 * 42; idx += 256) {
        int vi = idx / 42, ui = idx - vi * 42;
        int v = vmin + vi;
        float a = 0.0f;
        #pragma unroll
        for (int t = 0; t < 12; ++t) {
            int jv = v + t - 9;
            if (jv >= 0 && jv < 132 && ((jv & 1) == 0))
                a = fmaf(fu[t], st1[(jv >> 1) - q0][ui], a);
        }
        float l = (a >= 0.0f) ? a : a * 0.2f;   // leaky relu
        l *= 1.4142135623730951f;               // GAIN
        l = fminf(fmaxf(l, -256.0f), 256.0f);   // CLAMP
        sz[vi][ui] = l;
    }
    __syncthreads();

    for (int idx = tid; idx < 42 * 16; idx += 256) {
        int vi = idx / 16, j = idx - vi * 16;
        float a = 0.0f;
        #pragma unroll
        for (int t = 0; t < 12; ++t) a = fmaf(fd[t], sz[vi][2 * j + t], a);
        sA[vi][j] = a;
    }
    __syncthreads();

    {
        int il = tid >> 4, jl = tid & 15;
        float a = 0.0f;
        #pragma unroll
        for (int t = 0; t < 12; ++t) a = fmaf(fd[t], sA[2 * il + t][jl], a);
        out[((size_t)bc * 64 + (i0 + il)) * 64 + (j0 + jl)] = a;
    }
}

// ---------------------------------------------------------------------------
extern "C" void kernel_launch(void* const* d_in, const int* in_sizes, int n_in,
                              void* d_out, int out_size, void* d_ws, size_t ws_size,
                              hipStream_t stream) {
    const float* x    = (const float*)d_in[0];
    const float* w    = (const float*)d_in[1];
    const float* affw = (const float*)d_in[2];
    const float* affb = (const float*)d_in[3];
    const float* cw   = (const float*)d_in[4];
    const float* bias = (const float*)d_in[5];
    const float* upf  = (const float*)d_in[6];
    const float* dnf  = (const float*)d_in[7];
    const float* ema  = (const float*)d_in[8];
    float* out = (float*)d_out;
    char*  ws  = (char*)d_ws;

    float*          sbuf  = (float*)(ws + OFF_S);
    float*          wsq   = (float*)(ws + OFF_WSQ);
    float*          dcoef = (float*)(ws + OFF_DCOEF);
    unsigned short* wbf   = (unsigned short*)(ws + OFF_WBF);
    unsigned short* xbf   = (unsigned short*)(ws + OFF_XBF);
    float*          yconv = (float*)(ws + OFF_YCONV);

    style_kernel<<<16, 256, 0, stream>>>(w, affw, affb, sbuf);
    wsq_kernel<<<1024, 256, 0, stream>>>(cw, wsq);
    demod_kernel<<<16, 256, 0, stream>>>(sbuf, wsq, ema, dcoef);
    modw_kernel<<<73728, 256, 0, stream>>>(cw, sbuf, dcoef, wbf);
    xcvt_kernel<<<dim3(512, 8), 256, 0, stream>>>(x, xbf);
    conv_wmma_kernel<<<dim3((NTOT + NTILE - 1) / NTILE, COUT / MTILE, BATCH),
                       256, 0, stream>>>(wbf, xbf, bias, yconv);
    filter_fused_kernel<<<dim3(4, 4, BATCH * COUT), 256, 0, stream>>>(
        yconv, upf, dnf, out);
}
